// MHA_CrossAttention_40673340293163
// MI455X (gfx1250) — matrix operand order repaired and checked by reading, not compiled, not run on hardware
//
#include <hip/hip_runtime.h>
#include <hip/hip_bf16.h>

typedef __attribute__((ext_vector_type(16))) _Float16 v16h;
typedef __attribute__((ext_vector_type(8)))  _Float16 v8h;
typedef __attribute__((ext_vector_type(8)))  float    v8f;

union V16h { v16h v; v8h h[2]; };

#define NB   4
#define NC   256
#define LPIX 2304     // 48*48
#define NH   8
#define HD   32
#define GRID 48
#define MTILES (LPIX/16)   // 144

// ---------------------------------------------------------------- f32 -> f16
__global__ void cvt_f16_kernel(const float* __restrict__ src,
                               _Float16* __restrict__ dst, int n) {
    int i = blockIdx.x * blockDim.x + threadIdx.x;
    if (i < n) dst[i] = (_Float16)src[i];
}

// ------------------------------------------------------- LayerNorm pass 1: stats
// one thread per pixel; loop over 256 channels (coalesced in pixel dim)
__global__ void ln_stats_kernel(const float* __restrict__ x,
                                float* __restrict__ mean,
                                float* __restrict__ rstd) {
    int p = blockIdx.x * blockDim.x + threadIdx.x;   // b*LPIX + l
    if (p >= NB * LPIX) return;
    int b = p / LPIX, l = p % LPIX;
    float s = 0.f, sq = 0.f;
    for (int c = 0; c < NC; ++c) {
        float v = x[((size_t)b * NC + c) * LPIX + l];
        s += v; sq += v * v;
    }
    float mu = s * (1.0f / NC);
    float var = sq * (1.0f / NC) - mu * mu;
    mean[p] = mu;
    rstd[p] = rsqrtf(var + 1e-6f);
}

// -------------------------------------------------- LayerNorm pass 2: normalize
// reads NCHW fp32 coalesced, writes pixel-major f16 [b][l][c]
__global__ void ln_norm_kernel(const float* __restrict__ x,
                               const float* __restrict__ mean,
                               const float* __restrict__ rstd,
                               const float* __restrict__ w,
                               const float* __restrict__ bb,
                               _Float16* __restrict__ out) {
    size_t i = (size_t)blockIdx.x * blockDim.x + threadIdx.x;
    if (i >= (size_t)NB * NC * LPIX) return;
    int b = (int)(i / ((size_t)NC * LPIX));
    int r = (int)(i % ((size_t)NC * LPIX));
    int c = r / LPIX, l = r % LPIX;
    int p = b * LPIX + l;
    float v = (x[i] - mean[p]) * rstd[p] * w[c] + bb[c];
    out[((size_t)p) * NC + c] = (_Float16)v;
}

// --------------------------------------------------------- QKV GEMM (f16 WMMA)
// D[16 pix x 16 out] = sum_c A(act pixel-major) x B(W row-major [o][c])
// trans_out==0 : out[b][l][o] (pixel-major, for Q/K)
// trans_out==1 : out[b][o][l] (channel-major, for V)
__global__ __launch_bounds__(128)
void gemm_qkv_kernel(const _Float16* __restrict__ act,
                     const _Float16* __restrict__ W,
                     const float* __restrict__ bias,
                     _Float16* __restrict__ out, int trans_out) {
    int wave = threadIdx.x >> 5, lane = threadIdx.x & 31;
    int tid = blockIdx.x * 4 + wave;                 // 0 .. NB*MTILES*16-1
    int b  = tid / (MTILES * 16);
    int r  = tid % (MTILES * 16);
    int mt = r >> 4, nt = r & 15;
    int ln = lane & 15, hl = lane >> 4;

    size_t arow = ((size_t)b * LPIX + mt * 16 + ln) * NC;   // A row = pixel
    int o = nt * 16 + ln;                                   // B col = out chan
    const _Float16* wrow = W + (size_t)o * NC;

    v8f acc = {};
    #pragma unroll
    for (int c0 = 0; c0 < NC; c0 += 32) {
        V16h A, Bm;
        A.h[0] = *(const v8h*)(act + arow + c0 + 8 * hl);
        A.h[1] = *(const v8h*)(act + arow + c0 + 16 + 8 * hl);
        Bm.h[0] = *(const v8h*)(wrow + c0 + 8 * hl);
        Bm.h[1] = *(const v8h*)(wrow + c0 + 16 + 8 * hl);
        acc = __builtin_amdgcn_wmma_f32_16x16x32_f16(
            false, A.v, false, Bm.v, (short)0, acc, false, false);
    }
    float bv = bias[o];
    if (trans_out) {
        v8h ov;
        #pragma unroll
        for (int j = 0; j < 8; ++j) ov[j] = (_Float16)(acc[j] + bv);
        size_t base = ((size_t)b * NC + o) * LPIX + mt * 16 + 8 * hl;
        *(v8h*)(out + base) = ov;
    } else {
        #pragma unroll
        for (int j = 0; j < 8; ++j) {
            int p = mt * 16 + j + 8 * hl;
            out[((size_t)b * LPIX + p) * NC + o] = (_Float16)(acc[j] + bv);
        }
    }
}

// ------------------------------------------------ flash attention (f16 WMMA)
// one wave per (b, head, 16-query tile); 32 keys per step
__global__ __launch_bounds__(128)
void attn_kernel(const _Float16* __restrict__ q,
                 const _Float16* __restrict__ kmat,
                 const _Float16* __restrict__ vt,
                 const float* __restrict__ rel,
                 _Float16* __restrict__ aout) {
    __shared__ __align__(16) _Float16 pbuf[4][16 * 32];
    int wave = threadIdx.x >> 5, lane = threadIdx.x & 31;
    int tid = blockIdx.x * 4 + wave;                 // 0 .. NB*NH*MTILES-1
    int b  = tid / (NH * MTILES);
    int r  = tid % (NH * MTILES);
    int h  = r / MTILES;
    int qt = r % MTILES;
    int ln = lane & 15, hl = lane >> 4;
    _Float16* pb = pbuf[wave];

    // Q tile as WMMA A operand (M=query row, K=head dim 32)
    V16h aq;
    {
        size_t qrow = ((size_t)b * LPIX + qt * 16 + ln) * NC + h * HD;
        aq.h[0] = *(const v8h*)(q + qrow + 8 * hl);
        aq.h[1] = *(const v8h*)(q + qrow + 16 + 8 * hl);
    }
    int qy[8], qx[8];
    #pragma unroll
    for (int j = 0; j < 8; ++j) {
        int p = qt * 16 + j + 8 * hl;
        qy[j] = p / GRID; qx[j] = p % GRID;
    }

    float rm[8], rs[8];
    #pragma unroll
    for (int j = 0; j < 8; ++j) { rm[j] = -1e30f; rs[j] = 0.f; }
    v8f acc0 = {}, acc1 = {};
    const float scale = 0.17677669529663687f;        // 1/sqrt(32)

    for (int k0 = 0; k0 < LPIX; k0 += 32) {
        float s[2][8];
        #pragma unroll
        for (int t = 0; t < 2; ++t) {
            int key = k0 + 16 * t + ln;              // this lane's key column
            V16h Bk;                                 // B[d][k] = K[key][d]
            size_t krow = ((size_t)b * LPIX + key) * NC + h * HD + 8 * hl;
            Bk.h[0] = *(const v8h*)(kmat + krow);
            Bk.h[1] = *(const v8h*)(kmat + krow + 16);
            v8f zc = {};
            v8f st = __builtin_amdgcn_wmma_f32_16x16x32_f16(
                false, aq.v, false, Bk.v, (short)0, zc, false, false);
            int ky = key / GRID, kx = key % GRID;
            #pragma unroll
            for (int j = 0; j < 8; ++j) {
                int idx = (qy[j] - ky + 63) * 127 + (qx[j] - kx + 63);
                s[t][j] = st[j] * scale + rel[(size_t)idx * NH + h];
            }
        }
        // online softmax across the 32 keys (rows live in 16-lane halves)
        #pragma unroll
        for (int j = 0; j < 8; ++j) {
            float mx = fmaxf(s[0][j], s[1][j]);
            mx = fmaxf(mx, __shfl_xor(mx, 1));
            mx = fmaxf(mx, __shfl_xor(mx, 2));
            mx = fmaxf(mx, __shfl_xor(mx, 4));
            mx = fmaxf(mx, __shfl_xor(mx, 8));
            float nm = fmaxf(rm[j], mx);
            float sc = __expf(rm[j] - nm);
            rm[j] = nm;
            float p0 = __expf(s[0][j] - nm);
            float p1 = __expf(s[1][j] - nm);
            float ts = p0 + p1;
            ts += __shfl_xor(ts, 1);
            ts += __shfl_xor(ts, 2);
            ts += __shfl_xor(ts, 4);
            ts += __shfl_xor(ts, 8);
            rs[j] = rs[j] * sc + ts;
            acc0[j] *= sc; acc1[j] *= sc;
            int row = j + 8 * hl;                    // D layout -> LDS m-major
            pb[row * 32 + ln]      = (_Float16)p0;
            pb[row * 32 + ln + 16] = (_Float16)p1;
        }
        __syncthreads();
        // reload P in A-operand layout
        V16h Pa;
        Pa.h[0] = *(const v8h*)(pb + ln * 32 + 16 * hl);
        Pa.h[1] = *(const v8h*)(pb + ln * 32 + 16 * hl + 8);
        // V tiles as B operand: vt is [b][h*32+d][l] -> 16 contiguous keys/lane
        V16h Bv0, Bv1;
        {
            size_t vbase = ((size_t)b * NC + h * HD) * LPIX + k0 + 16 * hl;
            const _Float16* v0 = vt + vbase + (size_t)ln * LPIX;
            const _Float16* v1 = vt + vbase + (size_t)(ln + 16) * LPIX;
            Bv0.h[0] = *(const v8h*)(v0); Bv0.h[1] = *(const v8h*)(v0 + 8);
            Bv1.h[0] = *(const v8h*)(v1); Bv1.h[1] = *(const v8h*)(v1 + 8);
        }
        acc0 = __builtin_amdgcn_wmma_f32_16x16x32_f16(
            false, Pa.v, false, Bv0.v, (short)0, acc0, false, false);
        acc1 = __builtin_amdgcn_wmma_f32_16x16x32_f16(
            false, Pa.v, false, Bv1.v, (short)0, acc1, false, false);
        __syncthreads();
    }
    #pragma unroll
    for (int j = 0; j < 8; ++j) {
        float inv = 1.0f / rs[j];
        int p = qt * 16 + j + 8 * hl;
        size_t obase = ((size_t)b * LPIX + p) * NC + h * HD;
        aout[obase + ln]      = (_Float16)(acc0[j] * inv);
        aout[obase + ln + 16] = (_Float16)(acc1[j] * inv);
    }
}

// ---------------------------------------- projection GEMM + residual (WMMA)
__global__ __launch_bounds__(128)
void proj_kernel(const _Float16* __restrict__ act,
                 const _Float16* __restrict__ W,
                 const float* __restrict__ bias,
                 const float* __restrict__ xres,
                 float* __restrict__ out) {
    int wave = threadIdx.x >> 5, lane = threadIdx.x & 31;
    int tid = blockIdx.x * 4 + wave;
    int b  = tid / (MTILES * 16);
    int r  = tid % (MTILES * 16);
    int mt = r >> 4, nt = r & 15;
    int ln = lane & 15, hl = lane >> 4;

    size_t arow = ((size_t)b * LPIX + mt * 16 + ln) * NC;
    int o = nt * 16 + ln;                            // output channel
    const _Float16* wrow = W + (size_t)o * NC;

    v8f acc = {};
    #pragma unroll
    for (int c0 = 0; c0 < NC; c0 += 32) {
        V16h A, Bm;
        A.h[0] = *(const v8h*)(act + arow + c0 + 8 * hl);
        A.h[1] = *(const v8h*)(act + arow + c0 + 16 + 8 * hl);
        Bm.h[0] = *(const v8h*)(wrow + c0 + 8 * hl);
        Bm.h[1] = *(const v8h*)(wrow + c0 + 16 + 8 * hl);
        acc = __builtin_amdgcn_wmma_f32_16x16x32_f16(
            false, A.v, false, Bm.v, (short)0, acc, false, false);
    }
    float bv = bias[o];
    size_t base = ((size_t)b * NC + o) * LPIX + mt * 16 + 8 * hl;
    #pragma unroll
    for (int j = 0; j < 8; ++j)
        out[base + j] = acc[j] + bv + xres[base + j];
}

// ------------------------------------------------------------------- launcher
extern "C" void kernel_launch(void* const* d_in, const int* in_sizes, int n_in,
                              void* d_out, int out_size, void* d_ws, size_t ws_size,
                              hipStream_t stream) {
    const float* x      = (const float*)d_in[0];
    const float* ctx    = (const float*)d_in[1];
    const float* ln_x_w = (const float*)d_in[2];
    const float* ln_x_b = (const float*)d_in[3];
    const float* ln_c_w = (const float*)d_in[4];
    const float* ln_c_b = (const float*)d_in[5];
    const float* Wq     = (const float*)d_in[6];
    const float* bq     = (const float*)d_in[7];
    const float* Wk     = (const float*)d_in[8];
    const float* bk     = (const float*)d_in[9];
    const float* Wv     = (const float*)d_in[10];
    const float* bv     = (const float*)d_in[11];
    const float* Wp     = (const float*)d_in[12];
    const float* bp     = (const float*)d_in[13];
    const float* rel    = (const float*)d_in[14];
    float* out = (float*)d_out;

    size_t off = 0;
    auto alloc = [&](size_t bytes) -> void* {
        void* p = (char*)d_ws + off;
        off += (bytes + 255) & ~(size_t)255;
        return p;
    };
    const size_t WN = 256 * 256;              // weight elems
    const size_t AN = (size_t)NB * LPIX * NC; // activation elems
    _Float16* wq16 = (_Float16*)alloc(WN * 2);
    _Float16* wk16 = (_Float16*)alloc(WN * 2);
    _Float16* wv16 = (_Float16*)alloc(WN * 2);
    _Float16* wp16 = (_Float16*)alloc(WN * 2);
    float* meanx = (float*)alloc((size_t)NB * LPIX * 4);
    float* rstdx = (float*)alloc((size_t)NB * LPIX * 4);
    float* meanc = (float*)alloc((size_t)NB * LPIX * 4);
    float* rstdc = (float*)alloc((size_t)NB * LPIX * 4);
    _Float16* xn16 = (_Float16*)alloc(AN * 2);
    _Float16* cn16 = (_Float16*)alloc(AN * 2);
    _Float16* q16  = (_Float16*)alloc(AN * 2);
    _Float16* k16  = (_Float16*)alloc(AN * 2);
    _Float16* vt16 = (_Float16*)alloc(AN * 2);   // channel-major
    _Float16* ao16 = (_Float16*)alloc(AN * 2);

    // weights -> f16
    cvt_f16_kernel<<<(WN + 255) / 256, 256, 0, stream>>>(Wq, wq16, (int)WN);
    cvt_f16_kernel<<<(WN + 255) / 256, 256, 0, stream>>>(Wk, wk16, (int)WN);
    cvt_f16_kernel<<<(WN + 255) / 256, 256, 0, stream>>>(Wv, wv16, (int)WN);
    cvt_f16_kernel<<<(WN + 255) / 256, 256, 0, stream>>>(Wp, wp16, (int)WN);

    // layernorms
    int npix = NB * LPIX;
    ln_stats_kernel<<<(npix + 255) / 256, 256, 0, stream>>>(x, meanx, rstdx);
    ln_stats_kernel<<<(npix + 255) / 256, 256, 0, stream>>>(ctx, meanc, rstdc);
    ln_norm_kernel<<<(int)(AN / 256), 256, 0, stream>>>(x, meanx, rstdx, ln_x_w, ln_x_b, xn16);
    ln_norm_kernel<<<(int)(AN / 256), 256, 0, stream>>>(ctx, meanc, rstdc, ln_c_w, ln_c_b, cn16);

    // QKV projections (WMMA)
    int gtiles = NB * MTILES * 16;           // 9216 waves
    gemm_qkv_kernel<<<gtiles / 4, 128, 0, stream>>>(xn16, wq16, bq, q16, 0);
    gemm_qkv_kernel<<<gtiles / 4, 128, 0, stream>>>(cn16, wk16, bk, k16, 0);
    gemm_qkv_kernel<<<gtiles / 4, 128, 0, stream>>>(cn16, wv16, bv, vt16, 1);

    // flash attention (WMMA)
    int atiles = NB * NH * MTILES;           // 4608 waves
    attn_kernel<<<atiles / 4, 128, 0, stream>>>(q16, k16, vt16, rel, ao16);

    // projection + residual (WMMA)
    proj_kernel<<<gtiles / 4, 128, 0, stream>>>(ao16, wp16, bp, x, out);
}
